// LigandGraphEncoder_70050916597777
// MI455X (gfx1250) — compile-verified, matrix operand order checked by reading, not run on hardware
//
#include <hip/hip_runtime.h>
#include <hip/hip_bf16.h>

// ---------------------------------------------------------------------------
// LigandGraphEncoder for MI455X (gfx1250, wave32, WMMA + Tensor Data Mover).
// Heavy GEMMs: v_wmma_f32_16x16x32_bf16 with TDM (tensor_load_to_lds) A-tile
// staging, double-buffered LDS, TENSORcnt-tracked pipeline.
// ---------------------------------------------------------------------------

#define N_ATOMS 6000
#define NS_     20000
#define E_      60000
#define ET_     80000     // E + NS (self loops appended)
#define B_      16
#define INDIM_  78
#define EMB_    1024
#define HID_    2048
#define H_      4
#define C_      512

typedef __attribute__((ext_vector_type(16))) __bf16 v16bf;
typedef __attribute__((ext_vector_type(8)))  float  v8f;
typedef unsigned int uint32_tt;
typedef __attribute__((ext_vector_type(4)))  unsigned int v4u;
typedef __attribute__((ext_vector_type(4)))  int  v4i;
typedef __attribute__((ext_vector_type(8)))  int  v8i;

union Frag { uint4 q[2]; v16bf v; };

// ------------------------------ WMMA GEMM ----------------------------------
// C[M,N] = A[M,K](bf16, row-major) x B[K,N](bf16, row-major) (+ bias[N])
// Block tile 128x128, K-step 64, 256 threads = 8 waves in 2x4, wave tile 64x32.
// A tiles staged by the Tensor Data Mover (pad_enable gives the 160B LDS row
// pitch); B tiles prefetched to registers and transposed into LDS. Double
// buffered: one barrier per K-step, DMA + B-prefetch overlap the WMMAs.
#define BLK_M 128
#define BLK_N 128
#define BLK_K 64
#define LDSP  80      // bf16 pitch: 160B rows = 128B data + 32B TDM pad

// Issue a TDM 2D tile load: 128 rows x 64 bf16 from gptr (row stride K bf16)
// into LDS at lds_off. Element size 8B => row = 16 elems. rows_rem clamps the
// tensor so the DMA zero-fills rows past the end of A.
__device__ __forceinline__ void tdm_load_A(const __bf16* gptr, unsigned lds_off,
                                           int rows_rem, int K) {
  unsigned long long ga = (unsigned long long)(size_t)gptr;
  unsigned tdim0 = (unsigned)(K >> 2);      // tensor row length, 8B units
  unsigned tdim1 = (unsigned)rows_rem;      // rows remaining (OOB -> zero)
  unsigned tstr0 = (unsigned)(K >> 2);      // row stride, 8B units
  v4u g0;
  g0[0] = 1u;                                        // count=1, user mode
  g0[1] = lds_off;                                   // LDS byte address
  g0[2] = (unsigned)(ga & 0xffffffffu);              // global addr [31:0]
  g0[3] = (unsigned)((ga >> 32) & 0x01ffffffu) | 0x80000000u; // [56:32], type=2
  v8i g1;
  // data_size=3 (8B) | pad_enable | pad_interval=4 (32 dw) | pad_amount=7 (8 dw)
  g1[0] = (int)((3u << 16) | (1u << 20) | (4u << 22) | (7u << 25));
  g1[1] = (int)((tdim0 & 0xffffu) << 16);            // tensor_dim0 lo
  g1[2] = (int)((tdim0 >> 16) | ((tdim1 & 0xffffu) << 16));
  g1[3] = (int)((tdim1 >> 16) | (16u << 16));        // tile_dim0 = 16 elems
  g1[4] = (int)128u;                                 // tile_dim1 = 128 rows
  g1[5] = (int)tstr0;                                // tensor_dim0_stride lo
  g1[6] = 0;
  g1[7] = 0;
  v4i z4 = {0, 0, 0, 0};
#if defined(__clang_major__) && (__clang_major__ >= 23)
  v8i z8 = {0, 0, 0, 0, 0, 0, 0, 0};
  __builtin_amdgcn_tensor_load_to_lds(g0, g1, z4, z4, z8, 0);
#else
  __builtin_amdgcn_tensor_load_to_lds(g0, g1, z4, z4, 0);
#endif
}

__global__ __launch_bounds__(256) void k_gemm_bf16_wmma(
    const __bf16* __restrict__ A, const __bf16* __restrict__ B,
    float* __restrict__ Cm, const float* __restrict__ bias,
    int M, int K, int Nn)
{
  __shared__ __bf16 As[2][BLK_M][LDSP];
  __shared__ __bf16 Bs[2][BLK_N][LDSP];   // stored transposed: Bs[b][n][k]
  const int tid  = threadIdx.x;
  const int lane = tid & 31;
  const int wave = tid >> 5;
  const int wm = wave >> 2;            // 0..1 -> 64 rows
  const int wn = wave & 3;             // 0..3 -> 32 cols
  const int m0 = blockIdx.y * BLK_M;
  const int n0 = blockIdx.x * BLK_N;

  const int halfsel = lane >> 4;       // 0/1
  const int l16     = lane & 15;
  const int kbase   = halfsel * 8;

  v8f acc[4][2];
  #pragma unroll
  for (int i = 0; i < 4; ++i)
    #pragma unroll
    for (int j = 0; j < 2; ++j)
      #pragma unroll
      for (int v = 0; v < 8; ++v) acc[i][j][v] = 0.0f;

  const int nt = K / BLK_K;
  const int rows_rem = M - m0;         // TDM zero-fills rows >= rows_rem

  uint4 breg[4];
  const int bkrow = tid >> 4;          // 0..15 (+16,+32,+48 per chunk)
  const int bnb   = (tid & 15) * 8;

  // --- prologue: tile 0 ---
  if (wave == 0)
    tdm_load_A(A + (size_t)m0 * K, (unsigned)(size_t)&As[0][0][0], rows_rem, K);
  #pragma unroll
  for (int it = 0; it < 4; ++it)
    breg[it] = *(const uint4*)(B + (size_t)(bkrow + 16 * it) * Nn + n0 + bnb);
  #pragma unroll
  for (int it = 0; it < 4; ++it) {
    const __bf16* pv = (const __bf16*)&breg[it];
    #pragma unroll
    for (int j = 0; j < 8; ++j) Bs[0][bnb + j][bkrow + 16 * it] = pv[j];
  }
  if (wave == 0) __builtin_amdgcn_s_wait_tensorcnt(0);
  __syncthreads();

  for (int ti = 0; ti < nt; ++ti) {
    const int cur = ti & 1;
    const bool pre = (ti + 1) < nt;
    if (pre) {
      if (wave == 0)
        tdm_load_A(A + (size_t)m0 * K + (ti + 1) * BLK_K,
                   (unsigned)(size_t)&As[1 - cur][0][0], rows_rem, K);
      #pragma unroll
      for (int it = 0; it < 4; ++it)
        breg[it] = *(const uint4*)(B + (size_t)((ti + 1) * BLK_K + bkrow + 16 * it) * Nn
                                   + n0 + bnb);
    }

    // --- compute on buffer `cur` ---
    #pragma unroll
    for (int kk = 0; kk < BLK_K; kk += 32) {
      Frag a[4], b[2];
      #pragma unroll
      for (int i = 0; i < 4; ++i) {
        int r = wm * 64 + i * 16 + l16;
        a[i].q[0] = *(const uint4*)&As[cur][r][kk + kbase];
        a[i].q[1] = *(const uint4*)&As[cur][r][kk + kbase + 16];
      }
      #pragma unroll
      for (int j = 0; j < 2; ++j) {
        int cn = wn * 32 + j * 16 + l16;
        b[j].q[0] = *(const uint4*)&Bs[cur][cn][kk + kbase];
        b[j].q[1] = *(const uint4*)&Bs[cur][cn][kk + kbase + 16];
      }
      #pragma unroll
      for (int i = 0; i < 4; ++i)
        #pragma unroll
        for (int j = 0; j < 2; ++j)
          acc[i][j] = __builtin_amdgcn_wmma_f32_16x16x32_bf16(
              false, a[i].v, false, b[j].v, (short)0, acc[i][j], false, false);
    }

    if (pre) {
      #pragma unroll
      for (int it = 0; it < 4; ++it) {
        const __bf16* pv = (const __bf16*)&breg[it];
        #pragma unroll
        for (int j = 0; j < 8; ++j) Bs[1 - cur][bnb + j][bkrow + 16 * it] = pv[j];
      }
      if (wave == 0) __builtin_amdgcn_s_wait_tensorcnt(0);
    }
    __syncthreads();
  }

  // writeback (C frag: lanes 0-15 N=l16 M=v, lanes 16-31 N=l16 M=8+v)
  #pragma unroll
  for (int i = 0; i < 4; ++i)
    #pragma unroll
    for (int j = 0; j < 2; ++j) {
      int n  = n0 + wn * 32 + j * 16 + l16;
      float bv = bias ? bias[n] : 0.0f;
      #pragma unroll
      for (int v = 0; v < 8; ++v) {
        int m = m0 + wm * 64 + i * 16 + halfsel * 8 + v;
        if (m < M) Cm[(size_t)m * Nn + n] = acc[i][j][v] + bv;
      }
    }
}

// -------------------------- small / graph kernels --------------------------

__global__ void k_f2bf(const float* __restrict__ s, __bf16* __restrict__ d, int n) {
  int i = blockIdx.x * blockDim.x + threadIdx.x;
  if (i < n) d[i] = (__bf16)s[i];
}

// P[n,EMB] = x[n,78] @ Wn + bn (tiny, memory-bound)
__global__ __launch_bounds__(256) void k_node_proj(
    const float* __restrict__ x, const float* __restrict__ Wn,
    const float* __restrict__ bn, float* __restrict__ P)
{
  __shared__ float xs[INDIM_];
  int n = blockIdx.x;
  for (int i = threadIdx.x; i < INDIM_; i += 256) xs[i] = x[n * INDIM_ + i];
  __syncthreads();
  for (int o = threadIdx.x; o < EMB_; o += 256) {
    float s = bn[o];
    #pragma unroll 6
    for (int k = 0; k < INDIM_; ++k) s += xs[k] * Wn[k * EMB_ + o];
    P[(size_t)n * EMB_ + o] = s;
  }
}

__global__ __launch_bounds__(256) void k_gather_bf16(
    const float* __restrict__ P, const int* __restrict__ idx, __bf16* __restrict__ A)
{
  int j = blockIdx.x;
  int s = idx[j];
  for (int c = threadIdx.x; c < EMB_; c += 256)
    A[(size_t)j * EMB_ + c] = (__bf16)P[(size_t)s * EMB_ + c];
}

// CSR over destinations (edges + self loops), built once per launch
__global__ void k_deg(const int* __restrict__ eidx, int* __restrict__ deg) {
  int t = blockIdx.x * blockDim.x + threadIdx.x;
  if (t >= ET_) return;
  int d = (t < E_) ? eidx[E_ + t] : (t - E_);
  atomicAdd(&deg[d], 1);
}

__global__ __launch_bounds__(1024) void k_scan(
    const int* __restrict__ deg, int* __restrict__ rowptr, int* __restrict__ cursor)
{
  __shared__ int sums[1024];
  const int n = NS_;
  int tid = threadIdx.x;
  int per = (n + 1023) / 1024;
  int start = tid * per;
  int s = 0;
  for (int i = 0; i < per; ++i) { int id = start + i; if (id < n) s += deg[id]; }
  sums[tid] = s;
  __syncthreads();
  for (int off = 1; off < 1024; off <<= 1) {
    int v = (tid >= off) ? sums[tid - off] : 0;
    __syncthreads();
    sums[tid] += v;
    __syncthreads();
  }
  int base = (tid == 0) ? 0 : sums[tid - 1];
  for (int i = 0; i < per; ++i) {
    int id = start + i;
    if (id < n) { rowptr[id] = base; cursor[id] = base; base += deg[id]; }
  }
  if (tid == 0) rowptr[n] = sums[1023];
}

__global__ void k_fill(const int* __restrict__ eidx, int* __restrict__ cursor,
                       int* __restrict__ perm) {
  int t = blockIdx.x * blockDim.x + threadIdx.x;
  if (t >= ET_) return;
  int d = (t < E_) ? eidx[E_ + t] : (t - E_);
  int pos = atomicAdd(&cursor[d], 1);
  perm[pos] = t;
}

__global__ __launch_bounds__(256) void k_attr_mean(const float* __restrict__ attr,
                                                   float* __restrict__ consts) {
  __shared__ float red[256];
  int tid = threadIdx.x;
  float s = 0.f;
  for (int i = tid; i < E_; i += 256) s += attr[i];
  red[tid] = s;
  __syncthreads();
  for (int off = 128; off > 0; off >>= 1) { if (tid < off) red[tid] += red[tid + off]; __syncthreads(); }
  if (tid == 0) consts[0] = red[0] / (float)E_;
}

// s_edge[i,h] = attr[i]*c1[l,h] + c0[l,h]; c* from W_edge, att_edge, We, be
__global__ __launch_bounds__(256) void k_edge_consts(
    const float* __restrict__ W_edge, const float* __restrict__ att_edge,
    const float* __restrict__ We, const float* __restrict__ be,
    float* __restrict__ consts)
{
  __shared__ float r0[256], r1[256];
  int l = blockIdx.x / H_, h = blockIdx.x % H_;
  int tid = threadIdx.x;
  float a0 = 0.f, a1 = 0.f;
  const float* ae = att_edge + ((size_t)l * H_ + h) * C_;
  for (int k = tid; k < EMB_; k += 256) {
    const float* wrow = W_edge + ((size_t)l * EMB_ + k) * HID_ + h * C_;
    float u = 0.f;
    for (int c = 0; c < C_; ++c) u += wrow[c] * ae[c];
    a1 += We[k] * u;
    a0 += be[k] * u;
  }
  r0[tid] = a0; r1[tid] = a1;
  __syncthreads();
  for (int off = 128; off > 0; off >>= 1) {
    if (tid < off) { r0[tid] += r0[tid + off]; r1[tid] += r1[tid + off]; }
    __syncthreads();
  }
  if (tid == 0) { consts[1 + l * 8 + h] = r0[0]; consts[1 + l * 8 + 4 + h] = r1[0]; }
}

// one wave per node: s_src/s_dst dots, shfl reduction
__global__ __launch_bounds__(256) void k_scores(
    const float* __restrict__ xh, const float* __restrict__ att_src,
    const float* __restrict__ att_dst, float* __restrict__ ssrc,
    float* __restrict__ sdst, int l)
{
  int waveId = (blockIdx.x * blockDim.x + threadIdx.x) >> 5;
  int lane = threadIdx.x & 31;
  if (waveId >= NS_) return;
  const float* row = xh + (size_t)waveId * HID_;
  for (int h = 0; h < H_; ++h) {
    const float* as = att_src + ((size_t)l * H_ + h) * C_;
    const float* ad = att_dst + ((size_t)l * H_ + h) * C_;
    float s1 = 0.f, s2 = 0.f;
    for (int i = lane; i < C_; i += 32) {
      float v = row[h * C_ + i];
      s1 += v * as[i];
      s2 += v * ad[i];
    }
    for (int off = 16; off > 0; off >>= 1) {
      s1 += __shfl_down(s1, off, 32);
      s2 += __shfl_down(s2, off, 32);
    }
    if (lane == 0) { ssrc[waveId * H_ + h] = s1; sdst[waveId * H_ + h] = s2; }
  }
}

// per-destination segment softmax over CSR (deg ~4)
__global__ __launch_bounds__(256) void k_softmax(
    const int* __restrict__ rowptr, const int* __restrict__ perm,
    const int* __restrict__ eidx, const float* __restrict__ attr,
    const float* __restrict__ consts, const float* __restrict__ ssrc,
    const float* __restrict__ sdst, float* __restrict__ alpha, int l)
{
  int n = blockIdx.x * blockDim.x + threadIdx.x;
  if (n >= NS_) return;
  int beg = rowptr[n], end = rowptr[n + 1];
  float amean = consts[0];
  for (int h = 0; h < H_; ++h) {
    float c0 = consts[1 + l * 8 + h];
    float c1 = consts[1 + l * 8 + 4 + h];
    float sd = sdst[n * H_ + h];
    float m = -3.0e38f;
    for (int e = beg; e < end; ++e) {
      int t = perm[e];
      int s = (t < E_) ? eidx[t] : (t - E_);
      float av = (t < E_) ? attr[t] : amean;
      float lg = ssrc[s * H_ + h] + sd + c0 + c1 * av;
      lg = (lg > 0.f) ? lg : 0.2f * lg;
      m = fmaxf(m, lg);
    }
    float z = 0.f;
    for (int e = beg; e < end; ++e) {
      int t = perm[e];
      int s = (t < E_) ? eidx[t] : (t - E_);
      float av = (t < E_) ? attr[t] : amean;
      float lg = ssrc[s * H_ + h] + sd + c0 + c1 * av;
      lg = (lg > 0.f) ? lg : 0.2f * lg;
      float p = __expf(lg - m);
      alpha[(size_t)t * H_ + h] = p;
      z += p;
    }
    float inv = 1.f / (z + 1e-16f);
    for (int e = beg; e < end; ++e)
      alpha[(size_t)perm[e] * H_ + h] *= inv;
  }
}

// fused: message aggregation + conv_bias + LayerNorm + ReLU -> h (f32) + A (bf16)
__global__ __launch_bounds__(256) void k_aggregate_ln(
    const int* __restrict__ rowptr, const int* __restrict__ perm,
    const int* __restrict__ eidx, const float* __restrict__ xh,
    const float* __restrict__ alpha, const float* __restrict__ conv_bias,
    const float* __restrict__ gamma, const float* __restrict__ beta,
    float* __restrict__ hout, __bf16* __restrict__ abf, int l)
{
  __shared__ float red[256];
  int n = blockIdx.x;
  int tid = threadIdx.x;
  int beg = rowptr[n], end = rowptr[n + 1];
  float acc[8];
  #pragma unroll
  for (int i = 0; i < 8; ++i) acc[i] = 0.f;
  for (int e = beg; e < end; ++e) {
    int t = perm[e];
    int s = (t < E_) ? eidx[t] : (t - E_);
    float4 a4 = *(const float4*)(alpha + (size_t)t * H_);
    const float* xr = xh + (size_t)s * HID_;
    #pragma unroll
    for (int i = 0; i < 8; ++i) {
      int c = tid + 256 * i;
      float av = (c < C_) ? a4.x : (c < 2 * C_) ? a4.y : (c < 3 * C_) ? a4.z : a4.w;
      acc[i] += xr[c] * av;
    }
  }
  #pragma unroll
  for (int i = 0; i < 8; ++i) acc[i] += conv_bias[l * HID_ + tid + 256 * i];
  // mean
  float s1 = 0.f;
  #pragma unroll
  for (int i = 0; i < 8; ++i) s1 += acc[i];
  red[tid] = s1;
  __syncthreads();
  for (int off = 128; off > 0; off >>= 1) { if (tid < off) red[tid] += red[tid + off]; __syncthreads(); }
  float mu = red[0] / (float)HID_;
  __syncthreads();
  // var
  float s2 = 0.f;
  #pragma unroll
  for (int i = 0; i < 8; ++i) { float d = acc[i] - mu; s2 += d * d; }
  red[tid] = s2;
  __syncthreads();
  for (int off = 128; off > 0; off >>= 1) { if (tid < off) red[tid] += red[tid + off]; __syncthreads(); }
  float rstd = rsqrtf(red[0] / (float)HID_ + 1e-5f);
  #pragma unroll
  for (int i = 0; i < 8; ++i) {
    int c = tid + 256 * i;
    float v = (acc[i] - mu) * rstd * gamma[l * HID_ + c] + beta[l * HID_ + c];
    v = fmaxf(v, 0.f);
    hout[(size_t)n * HID_ + c] = v;
    abf[(size_t)n * HID_ + c] = (__bf16)v;
  }
}

__global__ void k_pool_sub(const float* __restrict__ h, const int* __restrict__ ind,
                           float* __restrict__ agg, float* __restrict__ cnt) {
  size_t gid = (size_t)blockIdx.x * blockDim.x + threadIdx.x;
  if (gid >= (size_t)NS_ * HID_) return;
  int j = (int)(gid / HID_);
  int c = (int)(gid % HID_);
  int d = ind[j];
  atomicAdd(&agg[(size_t)d * HID_ + c], h[gid]);
  if (c == 0) atomicAdd(&cnt[d], 1.f);
}

__global__ void k_fin_agg(const float* __restrict__ agg, const float* __restrict__ cnt,
                          __bf16* __restrict__ abf) {
  size_t gid = (size_t)blockIdx.x * blockDim.x + threadIdx.x;
  if (gid >= (size_t)N_ATOMS * HID_) return;
  int n = (int)(gid / HID_);
  abf[gid] = (__bf16)(agg[gid] / fmaxf(cnt[n], 1.f));
}

__global__ void k_pool_batch(const float* __restrict__ outn, const int* __restrict__ bidx,
                             float* __restrict__ out, float* __restrict__ gcnt) {
  size_t gid = (size_t)blockIdx.x * blockDim.x + threadIdx.x;
  if (gid >= (size_t)N_ATOMS * HID_) return;
  int n = (int)(gid / HID_);
  int c = (int)(gid % HID_);
  int b = bidx[n];
  atomicAdd(&out[(size_t)b * HID_ + c], outn[gid]);
  if (c == 0) atomicAdd(&gcnt[b], 1.f);
}

__global__ void k_div_out(float* __restrict__ out, const float* __restrict__ gcnt) {
  int gid = blockIdx.x * blockDim.x + threadIdx.x;
  if (gid >= B_ * HID_) return;
  out[gid] /= fmaxf(gcnt[gid / HID_], 1.f);
}

// ------------------------------- launcher ----------------------------------

extern "C" void kernel_launch(void* const* d_in, const int* in_sizes, int n_in,
                              void* d_out, int out_size, void* d_ws, size_t ws_size,
                              hipStream_t stream) {
  const float* x        = (const float*)d_in[0];
  const float* attr     = (const float*)d_in[1];
  const float* Wn       = (const float*)d_in[2];
  const float* bn       = (const float*)d_in[3];
  const float* We       = (const float*)d_in[4];
  const float* be       = (const float*)d_in[5];
  const float* W0       = (const float*)d_in[6];
  const float* Ws       = (const float*)d_in[7];
  const float* att_src  = (const float*)d_in[8];
  const float* att_dst  = (const float*)d_in[9];
  const float* att_edge = (const float*)d_in[10];
  const float* W_edge   = (const float*)d_in[11];
  const float* convb    = (const float*)d_in[12];
  const float* gamma    = (const float*)d_in[13];
  const float* beta     = (const float*)d_in[14];
  const float* Wf       = (const float*)d_in[15];
  const float* bfv      = (const float*)d_in[16];
  const int*   nidx     = (const int*)d_in[17];
  const int*   eidx     = (const int*)d_in[18];   // [2,E] row-major: src then dst
  const int*   sind     = (const int*)d_in[19];
  const int*   bidx     = (const int*)d_in[20];
  float* out = (float*)d_out;

  char* ws = (char*)d_ws;
  auto alloc = [&](size_t bytes) -> char* {
    char* p = ws;
    ws += (bytes + 255) & ~(size_t)255;
    return p;
  };
  float*  P      = (float*)alloc((size_t)N_ATOMS * EMB_ * 4);
  __bf16* A      = (__bf16*)alloc((size_t)NS_ * HID_ * 2);
  __bf16* W0b    = (__bf16*)alloc((size_t)EMB_ * HID_ * 2);
  __bf16* Ws0b   = (__bf16*)alloc((size_t)HID_ * HID_ * 2);
  __bf16* Ws1b   = (__bf16*)alloc((size_t)HID_ * HID_ * 2);
  __bf16* Wfb    = (__bf16*)alloc((size_t)HID_ * HID_ * 2);
  float*  xh     = (float*)alloc((size_t)NS_ * HID_ * 4);
  float*  hbuf   = (float*)alloc((size_t)NS_ * HID_ * 4);
  float*  ssrc   = (float*)alloc((size_t)NS_ * H_ * 4);
  float*  sdst   = (float*)alloc((size_t)NS_ * H_ * 4);
  float*  alpha  = (float*)alloc((size_t)ET_ * H_ * 4);
  float*  consts = (float*)alloc(64 * 4);
  int*    deg    = (int*)alloc((size_t)NS_ * 4);
  int*    rowptr = (int*)alloc((size_t)(NS_ + 1) * 4);
  int*    cursor = (int*)alloc((size_t)NS_ * 4);
  int*    perm   = (int*)alloc((size_t)ET_ * 4);
  float*  agg    = (float*)alloc((size_t)N_ATOMS * HID_ * 4);
  __bf16* aggbf  = (__bf16*)alloc((size_t)N_ATOMS * HID_ * 2);
  float*  cnt    = (float*)alloc((size_t)N_ATOMS * 4);
  float*  outn   = (float*)alloc((size_t)N_ATOMS * HID_ * 4);
  float*  gcnt   = (float*)alloc((size_t)B_ * 4);

  // zero accumulator buffers (graph-capture-safe async memsets)
  hipMemsetAsync(deg, 0, (size_t)NS_ * 4, stream);
  hipMemsetAsync(agg, 0, (size_t)N_ATOMS * HID_ * 4, stream);
  hipMemsetAsync(cnt, 0, (size_t)N_ATOMS * 4, stream);
  hipMemsetAsync(gcnt, 0, (size_t)B_ * 4, stream);
  hipMemsetAsync(out, 0, (size_t)B_ * HID_ * 4, stream);

  // weight conversion fp32 -> bf16
  {
    int n0c = EMB_ * HID_, n1c = HID_ * HID_;
    k_f2bf<<<(n0c + 255) / 256, 256, 0, stream>>>(W0, W0b, n0c);
    k_f2bf<<<(n1c + 255) / 256, 256, 0, stream>>>(Ws, Ws0b, n1c);
    k_f2bf<<<(n1c + 255) / 256, 256, 0, stream>>>(Ws + (size_t)HID_ * HID_, Ws1b, n1c);
    k_f2bf<<<(n1c + 255) / 256, 256, 0, stream>>>(Wf, Wfb, n1c);
  }

  // node projection + gather into subgraph node space (as bf16 GEMM operand)
  k_node_proj<<<N_ATOMS, 256, 0, stream>>>(x, Wn, bn, P);
  k_gather_bf16<<<NS_, 256, 0, stream>>>(P, nidx, A);

  // CSR over destinations (fixed for all 3 layers)
  k_deg<<<(ET_ + 255) / 256, 256, 0, stream>>>(eidx, deg);
  k_scan<<<1, 1024, 0, stream>>>(deg, rowptr, cursor);
  k_fill<<<(ET_ + 255) / 256, 256, 0, stream>>>(eidx, cursor, perm);

  // edge-score constants (collapses e @ W_edge to 2 scalars per layer/head)
  k_attr_mean<<<1, 256, 0, stream>>>(attr, consts);
  k_edge_consts<<<3 * H_, 256, 0, stream>>>(W_edge, att_edge, We, be, consts);

  dim3 gemmBlk(256);
  for (int l = 0; l < 3; ++l) {
    const __bf16* Wl = (l == 0) ? W0b : (l == 1 ? Ws0b : Ws1b);
    int K = (l == 0) ? EMB_ : HID_;
    dim3 grid(HID_ / BLK_N, (NS_ + BLK_M - 1) / BLK_M);
    k_gemm_bf16_wmma<<<grid, gemmBlk, 0, stream>>>(A, Wl, xh, nullptr, NS_, K, HID_);
    k_scores<<<(NS_ * 32 + 255) / 256, 256, 0, stream>>>(xh, att_src, att_dst, ssrc, sdst, l);
    k_softmax<<<(NS_ + 255) / 256, 256, 0, stream>>>(rowptr, perm, eidx, attr, consts,
                                                     ssrc, sdst, alpha, l);
    k_aggregate_ln<<<NS_, 256, 0, stream>>>(rowptr, perm, eidx, xh, alpha,
                                            convb, gamma, beta, hbuf, A, l);
  }

  // subgraph mean pooling -> atoms, then final projection (WMMA, fused bias)
  {
    size_t tot = (size_t)NS_ * HID_;
    k_pool_sub<<<(int)((tot + 255) / 256), 256, 0, stream>>>(hbuf, sind, agg, cnt);
    size_t tot2 = (size_t)N_ATOMS * HID_;
    k_fin_agg<<<(int)((tot2 + 255) / 256), 256, 0, stream>>>(agg, cnt, aggbf);
    dim3 grid(HID_ / BLK_N, (N_ATOMS + BLK_M - 1) / BLK_M);
    k_gemm_bf16_wmma<<<grid, gemmBlk, 0, stream>>>(aggbf, Wfb, outn, bfv,
                                                   N_ATOMS, HID_, HID_);
    k_pool_batch<<<(int)((tot2 + 255) / 256), 256, 0, stream>>>(outn, bidx, out, gcnt);
    k_div_out<<<(B_ * HID_ + 255) / 256, 256, 0, stream>>>(out, gcnt);
  }
}
